// SemanticAgent_19825569038894
// MI455X (gfx1250) — matrix operand order repaired and checked by reading, not compiled, not run on hardware
//
#include <hip/hip_runtime.h>
#include <hip/hip_bf16.h>
#include <math.h>

// Problem sizes (fixed by reference)
#define BATCH 2048
#define OBS   1024
#define NACT  512
#define NTYPE 64
#define EDIM  16
#define HDIM  64

typedef __attribute__((ext_vector_type(16))) _Float16 v16h;
typedef __attribute__((ext_vector_type(8)))  _Float16 v8h;
typedef __attribute__((ext_vector_type(8)))  float    v8f;

// ---------------------------------------------------------------------------
// Fast tanh: use CDNA5 native V_TANH_F32 trans op when the builtin exists;
// otherwise a branch-free exp-based form (v_exp_f32), never the ocml path.
// ---------------------------------------------------------------------------
__device__ __forceinline__ float fast_tanh(float x) {
#if __has_builtin(__builtin_amdgcn_tanh_f32)
    return __builtin_amdgcn_tanh_f32(x);
#elif __has_builtin(__builtin_amdgcn_tanhf)
    return __builtin_amdgcn_tanhf(x);
#else
    // tanh(x) = 1 - 2/(exp(2x)+1); inputs here are pre-activations (small),
    // and __expf saturates cleanly at the extremes.
    float e = __expf(x + x);
    return 1.0f - 2.0f / (e + 1.0f);
#endif
}

// ---------------------------------------------------------------------------
// Prep kernel 1: repack a row-major f32 weight [K,N] into f16 WMMA-B tiles.
// Layout: tile (kt,nt) of 32x16; per lane L: n = L&15, kbase = (L>>4)*8;
// halves j=0..15 hold W[kt*32 + kbase + (j&7) + (j>>3)*16][nt*16 + n].
// ---------------------------------------------------------------------------
__global__ void conv_b_kernel(const float* __restrict__ W, _Float16* __restrict__ out,
                              int K, int N) {
    int o = blockIdx.x * blockDim.x + threadIdx.x;
    int total = K * N;
    if (o >= total) return;
    int j    = o & 15;
    int lane = (o >> 4) & 31;
    int tile = o >> 9;
    int NT   = N >> 4;
    int nt   = tile % NT;
    int kt   = tile / NT;
    int n    = lane & 15;
    int kb   = (lane >> 4) * 8;
    int k    = kt * 32 + kb + (j & 7) + (j >> 3) * 16;
    out[o] = (_Float16)W[k * N + nt * 16 + n];
}

// ---------------------------------------------------------------------------
// Prep kernel 2: te[t][h] = ba[h] + sum_e emb[t][e] * We[e][h]   (64x64 f32)
// ---------------------------------------------------------------------------
__global__ void te_kernel(const float* __restrict__ emb, const float* __restrict__ We,
                          const float* __restrict__ ba, float* __restrict__ te) {
    int i = blockIdx.x * blockDim.x + threadIdx.x;
    if (i >= NTYPE * HDIM) return;
    int t = i >> 6, h = i & 63;
    float acc = ba[h];
#pragma unroll
    for (int e = 0; e < EDIM; ++e) acc += emb[t * EDIM + e] * We[e * HDIM + h];
    te[i] = acc;
}

// ---------------------------------------------------------------------------
// Main fused kernel: one wave32 per 16-row batch tile.
// ---------------------------------------------------------------------------
__global__ __launch_bounds__(32) void actor_critic_kernel(
    const float* __restrict__ x,
    const int*   __restrict__ action_types,
    const int*   __restrict__ action_counts,
    const int*   __restrict__ action,
    const _Float16* __restrict__ W1h,  const _Float16* __restrict__ Wc1h,
    const _Float16* __restrict__ W2h,  const _Float16* __restrict__ Wc2h,
    const _Float16* __restrict__ Wfh,
    const float* __restrict__ te_g,
    const float* __restrict__ b1,  const float* __restrict__ b2,
    const float* __restrict__ bc1, const float* __restrict__ bc2,
    const float* __restrict__ Wo,  const float* __restrict__ bo,
    const float* __restrict__ Wc3, const float* __restrict__ bc3,
    float* __restrict__ out) {

    __shared__ float    s_te[NTYPE * HDIM];   // 16 KB  type table (+ba folded in)
    __shared__ _Float16 s_h[16 * HDIM];       //  2 KB  A-operand staging (f16)
    __shared__ float    s_f[16 * HDIM];       //  4 KB  fWf (pre-activation)
    __shared__ float    s_s[16 * NTYPE];      //  4 KB  hc2 temp, then s[row][t]
    __shared__ int      s_at[16 * NACT];      // 32 KB  action_types tile
    __shared__ float    s_wo[HDIM];
    __shared__ float    s_wc3[HDIM];

    const int lane    = threadIdx.x;
    const int rowBase = blockIdx.x * 16;

    // ---- stage te via CDNA5 async global->LDS DMA (ASYNCcnt path) --------
#if defined(__gfx1250__)
    for (int i = lane * 4; i < NTYPE * HDIM; i += 32 * 4) {
        unsigned long long ga = (unsigned long long)(const void*)(te_g + i);
        unsigned int       la = (unsigned int)(unsigned long long)(const void*)(&s_te[i]);
        asm volatile("global_load_async_to_lds_b128 %0, %1, off"
                     :: "v"(la), "v"(ga) : "memory");
    }
#else
    for (int i = lane; i < NTYPE * HDIM; i += 32) s_te[i] = te_g[i];
#endif

    // action_types tile: one coalesced pass into LDS (read twice later)
    {
        const int* atp = action_types + (size_t)rowBase * NACT;
        for (int i = lane; i < 16 * NACT; i += 32) s_at[i] = atp[i];
    }
    for (int i = lane; i < HDIM; i += 32) { s_wo[i] = Wo[i]; s_wc3[i] = Wc3[i]; }

#if defined(__gfx1250__)
    asm volatile("s_wait_asynccnt 0x0" ::: "memory");
#endif
    __syncthreads();

    // ---------------- layer 1: [16,1024] x [1024,64]x2 via WMMA ------------
    v8f acc[8];
#pragma unroll
    for (int i = 0; i < 8; ++i) { v8f z = {}; acc[i] = z; }

    const int m  = lane & 15;          // A-row this lane feeds
    const int kb = (lane >> 4) * 8;    // A K sub-base per ISA layout
    const float* xrow = x + (size_t)(rowBase + m) * OBS;

    for (int kt = 0; kt < OBS / 32; ++kt) {
        const int k = kt * 32 + kb;
        __builtin_prefetch(xrow + k + 64, 0, 1);   // global_prefetch_b8
        v16h a;
#pragma unroll
        for (int j = 0; j < 8; ++j) {
            a[j]     = (_Float16)xrow[k + j];        // K = k+kb .. +7
            a[j + 8] = (_Float16)xrow[k + 16 + j];   // K = k+kb+16 .. +23
        }
#pragma unroll
        for (int nt = 0; nt < 4; ++nt) {
            v16h b = *(const v16h*)(W1h + (((kt * 4 + nt) * 32 + lane) << 4));
            acc[nt] = __builtin_amdgcn_wmma_f32_16x16x32_f16(
                false, a, false, b, (short)0, acc[nt], false, false);
        }
#pragma unroll
        for (int nt = 0; nt < 4; ++nt) {
            v16h b = *(const v16h*)(Wc1h + (((kt * 4 + nt) * 32 + lane) << 4));
            acc[4 + nt] = __builtin_amdgcn_wmma_f32_16x16x32_f16(
                false, a, false, b, (short)0, acc[4 + nt], false, false);
        }
    }

    // C layout: row = r + (lane>>4)*8, col = lane&15
    const int cr  = lane & 15;
    const int rup = (lane >> 4) << 3;

    // actor h1 = tanh(l1 + b1) -> s_h (f16, A staging)
#pragma unroll
    for (int nt = 0; nt < 4; ++nt) {
        float bias = b1[nt * 16 + cr];
#pragma unroll
        for (int r = 0; r < 8; ++r)
            s_h[(r + rup) * HDIM + nt * 16 + cr] = (_Float16)fast_tanh(acc[nt][r] + bias);
    }
    __syncthreads();

    // helper: read 16x32 f16 A fragment from s_h at K offset k2
#define LDS_A(k2, areg)                                            \
    {                                                              \
        const _Float16* p = &s_h[(lane & 15) * HDIM + (k2) + kb];  \
        v8h lo = *(const v8h*)p;                                   \
        v8h hi = *(const v8h*)(p + 16);                            \
        _Pragma("unroll")                                          \
        for (int j = 0; j < 8; ++j) { areg[j] = lo[j]; areg[j + 8] = hi[j]; } \
    }

    // ---------------- actor layer 2: feats = tanh(h1 @ W2 + b2) ------------
    v8f acc2[4];
#pragma unroll
    for (int i = 0; i < 4; ++i) { v8f z = {}; acc2[i] = z; }
#pragma unroll
    for (int kt = 0; kt < 2; ++kt) {
        v16h a; LDS_A(kt * 32, a);
#pragma unroll
        for (int nt = 0; nt < 4; ++nt) {
            v16h b = *(const v16h*)(W2h + (((kt * 4 + nt) * 32 + lane) << 4));
            acc2[nt] = __builtin_amdgcn_wmma_f32_16x16x32_f16(
                false, a, false, b, (short)0, acc2[nt], false, false);
        }
    }
    __syncthreads();
#pragma unroll
    for (int nt = 0; nt < 4; ++nt) {
        float bias = b2[nt * 16 + cr];
#pragma unroll
        for (int r = 0; r < 8; ++r)
            s_h[(r + rup) * HDIM + nt * 16 + cr] = (_Float16)fast_tanh(acc2[nt][r] + bias);
    }
    __syncthreads();

    // ---------------- fWf = feats @ Wf (no activation) ---------------------
    v8f acc3[4];
#pragma unroll
    for (int i = 0; i < 4; ++i) { v8f z = {}; acc3[i] = z; }
#pragma unroll
    for (int kt = 0; kt < 2; ++kt) {
        v16h a; LDS_A(kt * 32, a);
#pragma unroll
        for (int nt = 0; nt < 4; ++nt) {
            v16h b = *(const v16h*)(Wfh + (((kt * 4 + nt) * 32 + lane) << 4));
            acc3[nt] = __builtin_amdgcn_wmma_f32_16x16x32_f16(
                false, a, false, b, (short)0, acc3[nt], false, false);
        }
    }
#pragma unroll
    for (int nt = 0; nt < 4; ++nt)
#pragma unroll
        for (int r = 0; r < 8; ++r)
            s_f[(r + rup) * HDIM + nt * 16 + cr] = acc3[nt][r];
    __syncthreads();

    // ---------------- critic layer 2 ---------------------------------------
#pragma unroll
    for (int nt = 0; nt < 4; ++nt) {
        float bias = bc1[nt * 16 + cr];
#pragma unroll
        for (int r = 0; r < 8; ++r)
            s_h[(r + rup) * HDIM + nt * 16 + cr] = (_Float16)fast_tanh(acc[4 + nt][r] + bias);
    }
    __syncthreads();
    v8f acc4[4];
#pragma unroll
    for (int i = 0; i < 4; ++i) { v8f z = {}; acc4[i] = z; }
#pragma unroll
    for (int kt = 0; kt < 2; ++kt) {
        v16h a; LDS_A(kt * 32, a);
#pragma unroll
        for (int nt = 0; nt < 4; ++nt) {
            v16h b = *(const v16h*)(Wc2h + (((kt * 4 + nt) * 32 + lane) << 4));
            acc4[nt] = __builtin_amdgcn_wmma_f32_16x16x32_f16(
                false, a, false, b, (short)0, acc4[nt], false, false);
        }
    }
    // hc2 -> s_s (temp, f32)
#pragma unroll
    for (int nt = 0; nt < 4; ++nt) {
        float bias = bc2[nt * 16 + cr];
#pragma unroll
        for (int r = 0; r < 8; ++r)
            s_s[(r + rup) * HDIM + nt * 16 + cr] = fast_tanh(acc4[nt][r] + bias);
    }
    __syncthreads();

    // value per row (lanes 0..15)
    float value = 0.0f;
    if (lane < 16) {
        float v = bc3[0];
#pragma unroll 8
        for (int h = 0; h < HDIM; ++h) v += s_s[lane * HDIM + h] * s_wc3[h];
        value = v;
    }
    __syncthreads();

    // ---------------- s[row][t] table (overwrites s_s) ---------------------
    const float bo0 = bo[0];
    for (int i = lane; i < 16 * NTYPE; i += 32) {
        int row = i >> 6, t = i & 63;
        const float* fw = &s_f[row * HDIM];
        const float* tt = &s_te[t * HDIM];
        float sacc = bo0;
#pragma unroll 8
        for (int h = 0; h < HDIM; ++h) sacc += fast_tanh(fw[h] + tt[h]) * s_wo[h];
        s_s[i] = sacc;
    }
    __syncthreads();

    // ---------------- masked log-softmax / entropy (2 lanes per row) -------
    const int row  = lane >> 1;
    const int hf   = lane & 1;
    const int grow = rowBase + row;
    const int cnt  = action_counts[grow];
    const int* tloc = &s_at[row * NACT];
    const float* srow = &s_s[row * NTYPE];

    int a0 = hf * 256;
    int a1 = min(cnt, a0 + 256);

    float mmax = -3.0e38f;
    for (int a = a0; a < a1; ++a) mmax = fmaxf(mmax, srow[tloc[a]]);
    mmax = fmaxf(mmax, __shfl_xor(mmax, 1));

    float S1 = 0.0f, S2 = 0.0f;
    for (int a = a0; a < a1; ++a) {
        float d = srow[tloc[a]] - mmax;
        float e = __expf(d);
        S1 += e;
        S2 += e * d;
    }
    S1 += __shfl_xor(S1, 1);
    S2 += __shfl_xor(S2, 1);

    float val = __shfl(value, row);   // all lanes active for the shuffle
    if (hf == 0) {
        float logS1 = __logf(S1);
        int   act   = action[grow];
        float sca   = srow[tloc[act]];
        out[(size_t)grow * 3 + 0] = sca - mmax - logS1;   // log_prob
        out[(size_t)grow * 3 + 1] = logS1 - S2 / S1;      // entropy
        out[(size_t)grow * 3 + 2] = val;                  // value
    }
#undef LDS_A
}

// ---------------------------------------------------------------------------
extern "C" void kernel_launch(void* const* d_in, const int* in_sizes, int n_in,
                              void* d_out, int out_size, void* d_ws, size_t ws_size,
                              hipStream_t stream) {
    const float* x    = (const float*)d_in[0];
    const int*   at   = (const int*)d_in[1];
    const int*   acnt = (const int*)d_in[2];
    const int*   act  = (const int*)d_in[3];
    const float* emb  = (const float*)d_in[4];
    const float* W1   = (const float*)d_in[5];
    const float* b1   = (const float*)d_in[6];
    const float* W2   = (const float*)d_in[7];
    const float* b2   = (const float*)d_in[8];
    const float* Wf   = (const float*)d_in[9];
    const float* We   = (const float*)d_in[10];
    const float* ba   = (const float*)d_in[11];
    const float* Wo   = (const float*)d_in[12];
    const float* bo   = (const float*)d_in[13];
    const float* Wc1  = (const float*)d_in[14];
    const float* bc1  = (const float*)d_in[15];
    const float* Wc2  = (const float*)d_in[16];
    const float* bc2  = (const float*)d_in[17];
    const float* Wc3  = (const float*)d_in[18];
    const float* bc3  = (const float*)d_in[19];

    // workspace layout (halves then floats)
    _Float16* W1h  = (_Float16*)d_ws;
    _Float16* Wc1h = W1h  + OBS * HDIM;       // 65536
    _Float16* W2h  = Wc1h + OBS * HDIM;       // 65536
    _Float16* Wc2h = W2h  + HDIM * HDIM;      // 4096
    _Float16* Wfh  = Wc2h + HDIM * HDIM;      // 4096
    float*    te   = (float*)(Wfh + HDIM * HDIM); // byte offset 286720, 16B aligned

    conv_b_kernel<<<(OBS * HDIM + 255) / 256, 256, 0, stream>>>(W1,  W1h,  OBS,  HDIM);
    conv_b_kernel<<<(OBS * HDIM + 255) / 256, 256, 0, stream>>>(Wc1, Wc1h, OBS,  HDIM);
    conv_b_kernel<<<(HDIM * HDIM + 255) / 256, 256, 0, stream>>>(W2,  W2h,  HDIM, HDIM);
    conv_b_kernel<<<(HDIM * HDIM + 255) / 256, 256, 0, stream>>>(Wc2, Wc2h, HDIM, HDIM);
    conv_b_kernel<<<(HDIM * HDIM + 255) / 256, 256, 0, stream>>>(Wf,  Wfh,  HDIM, HDIM);
    te_kernel<<<(NTYPE * HDIM + 255) / 256, 256, 0, stream>>>(emb, We, ba, te);

    actor_critic_kernel<<<BATCH / 16, 32, 0, stream>>>(
        x, at, acnt, act, W1h, Wc1h, W2h, Wc2h, Wfh, te,
        b1, b2, bc1, bc2, Wo, bo, Wc3, bc3, (float*)d_out);
}
